// HeteroGATVerifier_463856468432
// MI455X (gfx1250) — matrix operand order repaired and checked by reading, not compiled, not run on hardware
//
#include <hip/hip_runtime.h>
#include <hip/hip_fp16.h>
#include <hip/hip_bf16.h>

typedef __attribute__((ext_vector_type(16))) _Float16 v16h;
typedef __attribute__((ext_vector_type(8)))  float    v8f;

#define NEG_SLOPE 0.2f

// swap bits 3 and 4 of a 0..31 index: maps K index -> swizzled A-frag position
__device__ __forceinline__ int swap34(int c) {
  return (c & 7) | ((c & 8) << 1) | ((c & 16) >> 1);
}

// ---------------------------------------------------------------------------
// One-time weight pack: f32 [K x 128] row-major -> f16 fragment-ready layout
//   Wh[((k>>5)*128 + c)*32 + (k&31)]  (per 32-K chunk, per column, contiguous K)
// ---------------------------------------------------------------------------
__global__ void pack_weights_kernel(const float* __restrict__ W, _Float16* __restrict__ Wh,
                                    int total /* = K*128 */) {
  int i = blockIdx.x * blockDim.x + threadIdx.x;
  if (i >= total) return;
  int k = i >> 7, c = i & 127;
  Wh[(((size_t)(k >> 5) * 128 + c) << 5) | (k & 31)] = (_Float16)W[i];
}

// ---------------------------------------------------------------------------
// WMMA GEMM: C[M x 128] = A[M x K](f32) * B[K x 128](pre-packed f16) (+bias)
// block = 128 threads (4 waves); each wave does 32 rows x 128 cols
// (2 A-frags, 16 v8f accumulators, 16 v_wmma per 32-K step).
// B fragments load straight from global (32B contiguous per lane, L2-resident).
// A staged in LDS in contiguous-K layout -> 1 contiguous 32B LDS read per lane.
// Optional fused epilogue: dotOut[m] = C_row[m] . avec (pre-bias).
// ---------------------------------------------------------------------------
__global__ __launch_bounds__(128)
void gemm128_wmma_kernel(const float* __restrict__ A, const _Float16* __restrict__ Bh,
                         float* __restrict__ C, int M, int K,
                         const float* __restrict__ bias,
                         const float* __restrict__ avec, float* __restrict__ dotOut) {
  __shared__ _Float16 sA[128 * 32];   // 8 KB
  const int wave = threadIdx.x >> 5;
  const int lane = threadIdx.x & 31;
  const int m16  = lane & 15;
  const int h16  = lane >> 4;
  const int rowBase = blockIdx.x * 128;

  v8f acc0[8], acc1[8];
  const v8f vz = {0.f, 0.f, 0.f, 0.f, 0.f, 0.f, 0.f, 0.f};
#pragma unroll
  for (int t = 0; t < 8; ++t) { acc0[t] = vz; acc1[t] = vz; }

  const int nChunks = K >> 5;
  for (int kc = 0; kc < nChunks; ++kc) {
    const int k0 = kc << 5;
    // stage A tile: 128 rows x 32 K, f32->f16 pairs, contiguous K positions
    for (int i = threadIdx.x; i < 128 * 16; i += 128) {
      int r = i >> 4, jp = i & 15;
      int gr = rowBase + r;
      float vx = 0.f, vy = 0.f;
      if (gr < M) {
        const float2 v2 = ((const float2*)(A + (size_t)gr * K + k0))[jp];
        vx = v2.x; vy = v2.y;
      }
      *(__half2*)(&sA[r * 32 + 2 * jp]) = __floats2half2_rn(vx, vy);
    }
    __syncthreads();

    const v16h a0 = *(const v16h*)(&sA[(wave * 32      + m16) * 32 + h16 * 16]);
    const v16h a1 = *(const v16h*)(&sA[(wave * 32 + 16 + m16) * 32 + h16 * 16]);

    const _Float16* Bc = Bh + ((size_t)kc * 128) * 32;
#pragma unroll
    for (int t = 0; t < 8; ++t) {
      const int col = t * 16 + m16;
      const v16h bf = *(const v16h*)(Bc + (size_t)col * 32 + h16 * 16);
      acc0[t] = __builtin_amdgcn_wmma_f32_16x16x32_f16(
          false, a0, false, bf, (short)0, acc0[t], false, false);
      acc1[t] = __builtin_amdgcn_wmma_f32_16x16x32_f16(
          false, a1, false, bf, (short)0, acc1[t], false, false);
    }
    __syncthreads();
  }

  // C layout per acc: VGPR v -> M = v + 8*h16, N = m16 (+ t*16)
  if (C) {
#pragma unroll
    for (int t = 0; t < 8; ++t) {
      const float bcol = bias ? bias[t * 16 + m16] : 0.0f;
#pragma unroll
      for (int v = 0; v < 8; ++v) {
        int gr0 = rowBase + wave * 32 + h16 * 8 + v;
        int gr1 = gr0 + 16;
        if (gr0 < M) C[(size_t)gr0 * 128 + t * 16 + m16] = acc0[t][v] + bcol;
        if (gr1 < M) C[(size_t)gr1 * 128 + t * 16 + m16] = acc1[t][v] + bcol;
      }
    }
  }

  // fused attention dot: dotOut[m] = sum_n acc[m][n] * avec[n]   (pre-bias)
  if (avec) {
    float av[8];
#pragma unroll
    for (int t = 0; t < 8; ++t) av[t] = avec[t * 16 + m16];
#pragma unroll
    for (int v = 0; v < 8; ++v) {
      float s0 = 0.f, s1 = 0.f;
#pragma unroll
      for (int t = 0; t < 8; ++t) { s0 += acc0[t][v] * av[t]; s1 += acc1[t][v] * av[t]; }
#pragma unroll
      for (int mk = 8; mk > 0; mk >>= 1) {   // reduce over the 16 lanes of this half
        s0 += __shfl_xor(s0, mk, 32);
        s1 += __shfl_xor(s1, mk, 32);
      }
      if (m16 == 0) {
        int gr0 = rowBase + wave * 32 + h16 * 8 + v;
        int gr1 = gr0 + 16;
        if (gr0 < M) dotOut[gr0] = s0;
        if (gr1 < M) dotOut[gr1] = s1;
      }
    }
  }
}

// ---------------------------------------------------------------------------
// row dot: out[n] = dot(X[n, 0:128], a) (+ bias[0]) -- one wave per row
// ---------------------------------------------------------------------------
__global__ __launch_bounds__(256)
void rowdot_kernel(const float* __restrict__ X, const float* __restrict__ a,
                   const float* __restrict__ bias, float* __restrict__ out, int N) {
  int wid  = blockIdx.x * (blockDim.x >> 5) + (threadIdx.x >> 5);
  int lane = threadIdx.x & 31;
  if (wid >= N) return;
  float4 x = ((const float4*)(X + (size_t)wid * 128))[lane];
  float4 w = ((const float4*)a)[lane];
  float s = x.x * w.x + x.y * w.y + x.z * w.z + x.w * w.w;
#pragma unroll
  for (int off = 16; off > 0; off >>= 1) s += __shfl_down(s, off, 32);
  if (lane == 0) out[wid] = s + (bias ? bias[0] : 0.0f);
}

// order-preserving float<->uint mapping for atomic max
__device__ __forceinline__ unsigned f2o(float f) {
  unsigned u = __float_as_uint(f);
  return (u & 0x80000000u) ? ~u : (u | 0x80000000u);
}
__device__ __forceinline__ float o2f(unsigned u) {
  return (u & 0x80000000u) ? __uint_as_float(u & 0x7FFFFFFFu) : __uint_as_float(~u);
}

__global__ void fill_u32_kernel(unsigned* p, unsigned v, int n) {
  int i = blockIdx.x * blockDim.x + threadIdx.x;
  if (i < n) p[i] = v;
}
__global__ void fill_f32_kernel(float* p, float v, int n) {
  int i = blockIdx.x * blockDim.x + threadIdx.x;
  if (i < n) p[i] = v;
}

__global__ void init_bias_kernel(float* __restrict__ out, const float* __restrict__ b1,
                                 const float* __restrict__ b2, int total) {
  int i = blockIdx.x * blockDim.x + threadIdx.x;
  if (i >= total) return;
  int c = i & 127;
  out[i] = b1[c] + (b2 ? b2[c] : 0.0f);
}

__global__ void relu_copy_kernel(float* __restrict__ dst, const float* __restrict__ src, int n) {
  int i = blockIdx.x * blockDim.x + threadIdx.x;
  if (i < n) { float v = src[i]; dst[i] = v > 0.0f ? v : 0.0f; }
}

// pass 1: e = leaky_relu(es[src]+ed[dst]); segment-max via ordered-uint atomicMax
__global__ void edge_score_kernel(const int* __restrict__ src, const int* __restrict__ dst,
                                  const float* __restrict__ es, const float* __restrict__ ed,
                                  float* __restrict__ e, unsigned* __restrict__ mU, int E) {
  int i = blockIdx.x * blockDim.x + threadIdx.x;
  if (i >= E) return;
  float v = es[src[i]] + ed[dst[i]];
  v = (v > 0.0f) ? v : v * NEG_SLOPE;
  e[i] = v;
  atomicMax(&mU[dst[i]], f2o(v));
}

// pass 2: ex = exp(e - m[dst]); denom += ex
__global__ void edge_exp_kernel(const int* __restrict__ dst, float* __restrict__ e,
                                const unsigned* __restrict__ mU, float* __restrict__ denom, int E) {
  int i = blockIdx.x * blockDim.x + threadIdx.x;
  if (i >= E) return;
  int d = dst[i];
  float ex = expf(e[i] - o2f(mU[d]));
  e[i] = ex;
  atomicAdd(&denom[d], ex);
}

// pass 3: out[dst] += alpha * hs[src]  (one wave per edge, 4 ch/lane; L2-resident atomics)
__global__ __launch_bounds__(256)
void edge_aggr_kernel(const int* __restrict__ src, const int* __restrict__ dst,
                      const float* __restrict__ e, const float* __restrict__ denom,
                      const float* __restrict__ hs, float* __restrict__ out, int E) {
  int edge = blockIdx.x * (blockDim.x >> 5) + (threadIdx.x >> 5);
  int lane = threadIdx.x & 31;
  if (edge >= E) return;
  int s = src[edge], d = dst[edge];
  float alpha = e[edge] / (denom[d] + 1e-16f);
  float4 h = ((const float4*)(hs + (size_t)s * 128))[lane];
  float* orow = out + (size_t)d * 128 + lane * 4;
  atomicAdd(orow + 0, alpha * h.x);
  atomicAdd(orow + 1, alpha * h.y);
  atomicAdd(orow + 2, alpha * h.z);
  atomicAdd(orow + 3, alpha * h.w);
}

static inline int cdiv(int a, int b) { return (a + b - 1) / b; }

extern "C" void kernel_launch(void* const* d_in, const int* in_sizes, int n_in,
                              void* d_out, int out_size, void* d_ws, size_t ws_size,
                              hipStream_t stream) {
  // ---- inputs --------------------------------------------------------------
  const float* xin[4];
  int Nn[4];
  for (int t = 0; t < 4; ++t) { xin[t] = (const float*)d_in[t]; Nn[t] = in_sizes[t] / 256; }
  const int* eraw[5]; int Ecnt[5];
  for (int r = 0; r < 5; ++r) { eraw[r] = (const int*)d_in[4 + r]; Ecnt[r] = in_sizes[4 + r] / 2; }
  const float* lin_W  = (const float*)d_in[9];   // (4,256,128)
  const float* lin_b  = (const float*)d_in[10];  // (4,128)
  const float* convWs = (const float*)d_in[11];  // (2,5,128,128)
  const float* convWd = (const float*)d_in[12];
  const float* convAs = (const float*)d_in[13];  // (2,5,128)
  const float* convAd = (const float*)d_in[14];
  const float* convB  = (const float*)d_in[15];
  const float* outW   = (const float*)d_in[16];  // (128,1)
  const float* outB   = (const float*)d_in[17];  // (1,)

  // relations: src,dst node-type indices (QENT=0, CENT=1, SPAN=2, SENT=3)
  const int RS[5] = {0, 0, 1, 2, 1};
  const int RD[5] = {2, 3, 3, 1, 1};

  int Nmax = 0, Emax = 0;
  for (int t = 0; t < 4; ++t) Nmax = Nn[t] > Nmax ? Nn[t] : Nmax;
  for (int r = 0; r < 5; ++r) Emax = Ecnt[r] > Emax ? Ecnt[r] : Emax;

  // ---- workspace partition -------------------------------------------------
  float* ws = (float*)d_ws;
  size_t off = 0;
  float* X[4];
  for (int t = 0; t < 4; ++t) { X[t] = ws + off; off += (size_t)Nn[t] * 128; }
  float* XN[4] = {nullptr, nullptr, nullptr, nullptr};
  for (int t = 1; t < 4; ++t) { XN[t] = ws + off; off += (size_t)Nn[t] * 128; }
  float* H   = ws + off; off += (size_t)Nmax * 128;      // hs buffer
  float* Sv  = ws + off; off += (size_t)Nmax;            // es per src node
  float* Dv  = ws + off; off += (size_t)Nmax;            // ed per dst node
  unsigned* MU = (unsigned*)(ws + off); off += (size_t)Nmax;
  float* DEN = ws + off; off += (size_t)Nmax;
  float* EE  = ws + off; off += (size_t)Emax;            // per-edge score / exp
  off = (off + 7) & ~(size_t)7;                          // 32B align f16 blocks
  _Float16* WhLin = (_Float16*)(ws + off); off += (size_t)(4 * 256 * 128) / 2;
  _Float16* WhS   = (_Float16*)(ws + off); off += (size_t)(10 * 128 * 128) / 2;
  _Float16* WhD   = (_Float16*)(ws + off); off += (size_t)(10 * 128 * 128) / 2;

  // ---- pack all weights to f16 fragment layout (once per call) -------------
  for (int t = 0; t < 4; ++t)
    pack_weights_kernel<<<cdiv(256 * 128, 256), 256, 0, stream>>>(
        lin_W + (size_t)t * 256 * 128, WhLin + (size_t)t * 256 * 128, 256 * 128);
  for (int i = 0; i < 10; ++i) {
    pack_weights_kernel<<<cdiv(128 * 128, 256), 256, 0, stream>>>(
        convWs + (size_t)i * 128 * 128, WhS + (size_t)i * 128 * 128, 128 * 128);
    pack_weights_kernel<<<cdiv(128 * 128, 256), 256, 0, stream>>>(
        convWd + (size_t)i * 128 * 128, WhD + (size_t)i * 128 * 128, 128 * 128);
  }

  // ---- input projections: X[t] = xin[t] @ lin_W[t] + lin_b[t] --------------
  for (int t = 0; t < 4; ++t)
    gemm128_wmma_kernel<<<cdiv(Nn[t], 128), 128, 0, stream>>>(
        xin[t], WhLin + (size_t)t * 256 * 128, X[t], Nn[t], 256,
        lin_b + t * 128, nullptr, nullptr);

  // ---- 2 GAT layers --------------------------------------------------------
  for (int l = 0; l < 2; ++l) {
    init_bias_kernel<<<cdiv(Nn[2] * 128, 256), 256, 0, stream>>>(
        XN[2], convB + (size_t)(l * 5 + 0) * 128, nullptr, Nn[2] * 128);
    init_bias_kernel<<<cdiv(Nn[3] * 128, 256), 256, 0, stream>>>(
        XN[3], convB + (size_t)(l * 5 + 1) * 128, convB + (size_t)(l * 5 + 2) * 128, Nn[3] * 128);
    init_bias_kernel<<<cdiv(Nn[1] * 128, 256), 256, 0, stream>>>(
        XN[1], convB + (size_t)(l * 5 + 3) * 128, convB + (size_t)(l * 5 + 4) * 128, Nn[1] * 128);

    for (int r = 0; r < 5; ++r) {
      const int s = RS[r], d = RD[r];
      const int E = Ecnt[r];
      const int* esrc = eraw[r];
      const int* edst = eraw[r] + E;
      const size_t wo = (size_t)(l * 5 + r);

      // ed = (X[d] @ Wd) . a_d  -- dot fused into GEMM epilogue, C not stored
      gemm128_wmma_kernel<<<cdiv(Nn[d], 128), 128, 0, stream>>>(
          X[d], WhD + wo * 128 * 128, nullptr, Nn[d], 128,
          nullptr, convAd + wo * 128, Dv);

      // hs = X[s] @ Ws (stored, needed for aggregation); es = hs . a_s fused
      gemm128_wmma_kernel<<<cdiv(Nn[s], 128), 128, 0, stream>>>(
          X[s], WhS + wo * 128 * 128, H, Nn[s], 128,
          nullptr, convAs + wo * 128, Sv);

      // segment softmax + weighted scatter
      fill_u32_kernel<<<cdiv(Nn[d], 256), 256, 0, stream>>>(MU, 0u, Nn[d]);
      edge_score_kernel<<<cdiv(E, 256), 256, 0, stream>>>(esrc, edst, Sv, Dv, EE, MU, E);
      fill_f32_kernel<<<cdiv(Nn[d], 256), 256, 0, stream>>>(DEN, 0.0f, Nn[d]);
      edge_exp_kernel<<<cdiv(E, 256), 256, 0, stream>>>(edst, EE, MU, DEN, E);
      edge_aggr_kernel<<<cdiv(E, 8), 256, 0, stream>>>(esrc, edst, EE, DEN, H, XN[d], E);
    }

    for (int t = 1; t < 4; ++t)
      relu_copy_kernel<<<cdiv(Nn[t] * 128, 256), 256, 0, stream>>>(X[t], XN[t], Nn[t] * 128);
  }

  // ---- final head: out = X[SPAN] @ out_W + out_b ---------------------------
  rowdot_kernel<<<cdiv(Nn[2], 8), 256, 0, stream>>>(X[2], outW, outB, (float*)d_out, Nn[2]);
}
